// MultiHeadedAttention_2362232013394
// MI455X (gfx1250) — matrix operand order repaired and checked by reading, not compiled, not run on hardware
//
#include <hip/hip_runtime.h>

#define HEADS 16
#define DM    1024
#define DKH   64
#define HW    1024
#define BB    4
#define LQn   1024

typedef __attribute__((ext_vector_type(16))) __bf16 v16bf;
typedef __attribute__((ext_vector_type(8)))  float  v8f;
typedef __attribute__((ext_vector_type(4)))  unsigned int v4u;
typedef __attribute__((ext_vector_type(8)))  int v8i;
typedef __attribute__((ext_vector_type(4)))  int v4i;
typedef unsigned short u16t;
typedef unsigned int   u32t;

#if __has_builtin(__builtin_amdgcn_tensor_load_to_lds)
#define USE_TDM 1
#else
#define USE_TDM 0
#endif

union FragBF {
  v16bf v;
  u16t  u[16];
  u32t  d[8];
  uint4 q[2];
};

union U4 {
  uint4 q;
  u16t  us[8];
};

__device__ __forceinline__ u16t f2bf(float f) {
  u32t u = __float_as_uint(f);
  u32t r = u + 0x7fffu + ((u >> 16) & 1u);   // round-to-nearest-even
  return (u16t)(r >> 16);
}

#if USE_TDM
// Issue a 2D TDM tile load: global (row-major, stride0 elements/row) -> LDS.
// g1w0 packs data_size + pad_enable/pad_interval/pad_amount (D# group1 word 0).
// Groups 2/3 zero => 2D tensor (tile_dim2/3/4 = 0 -> unused).
__device__ __forceinline__ void tdm_load_2d(unsigned lds_off, const void* gptr,
                                            unsigned g1w0, int tensor_d0,
                                            int tensor_d1, int tile_d0,
                                            int tile_d1, int stride0) {
  unsigned long long ga = (unsigned long long)(size_t)gptr;
  v4u g0 = {1u,  // count=1, is_restore=0, gather off
            lds_off,
            (unsigned)ga,
            (unsigned)((ga >> 32) & 0x01FFFFFFu) | (2u << 30)};  // type=2
  v8i g1 = {(int)g1w0,
            (int)(((unsigned)tensor_d0 & 0xFFFFu) << 16),
            (int)((((unsigned)tensor_d0 >> 16) & 0xFFFFu) |
                  (((unsigned)tensor_d1 & 0xFFFFu) << 16)),
            (int)((((unsigned)tensor_d1 >> 16) & 0xFFFFu) |
                  ((unsigned)tile_d0 << 16)),
            tile_d1,      // tile_dim1 [143:128]; tile_dim2=0
            stride0,      // tensor_dim0_stride[31:0]
            0, 0};        // stride hi + tensor_dim1_stride (unused, 2D)
  v4i z = {0, 0, 0, 0};
#if defined(__clang_major__) && (__clang_major__ >= 23)
  v8i z8 = {0, 0, 0, 0, 0, 0, 0, 0};
  __builtin_amdgcn_tensor_load_to_lds(g0, g1, z, z, z8, 0);
#else
  __builtin_amdgcn_tensor_load_to_lds(g0, g1, z, z, 0);
#endif
}
// group1 word0 presets: data_size<<16 | pad_en<<20 | pad_int<<22 | pad_amt<<25
#define G1W0_A ((1u << 16) | (1u << 20) | (3u << 22) | (3u << 25))  // 2B, 16dw rows, +4dw pad -> stride 40 halves
#define G1W0_V ((1u << 16) | (1u << 20) | (4u << 22) | (3u << 25))  // 2B, 32dw rows, +4dw pad -> stride 72 halves
#define G1W0_M (2u << 16)                                           // 4B, no pad
#endif

// ---------------------------------------------------------------- f32 -> bf16
__global__ void cvt_f32_bf16(const float* __restrict__ in,
                             u16t* __restrict__ out, int n) {
  int i = (blockIdx.x * blockDim.x + threadIdx.x) * 4;
  if (i + 3 < n) {
    float4 f = *(const float4*)(in + i);
    out[i + 0] = f2bf(f.x);
    out[i + 1] = f2bf(f.y);
    out[i + 2] = f2bf(f.z);
    out[i + 3] = f2bf(f.w);
  }
}

// ---------------------------------------------------------------- GEMM (bf16)
// C(MxN) = A(MxK) @ B(KxN) + bias ; A,B bf16 row-major.
// Block: 128 threads (4 waves), tile 64(M) x 128(N); wave tile 32x64.
// A tile: TDM DMA into LDS [m][k] (stride 40 via TDM padding).
// B tile: manual transposed stage [n][k] (stride 40) -> B frags = 2x b128.
__global__ __launch_bounds__(128)
void gemm_bf16(const u16t* __restrict__ A, const u16t* __restrict__ Bm,
               const float* __restrict__ bias, u16t* __restrict__ Cbf,
               float* __restrict__ Cf, int M, int N, int K) {
  __shared__ u16t As[64 * 40];
  __shared__ u16t BsT[128 * 40];

  const int tid  = threadIdx.x;
  const int lane = tid & 31;
  const int hf   = lane >> 4;
  const int ln   = lane & 15;
  const int wave = tid >> 5;
  const int wm   = wave >> 1, wn = wave & 1;
  const int mbase = blockIdx.y * 64;
  const int nbase = blockIdx.x * 128;

  v8f acc[2][4];
#pragma unroll
  for (int a = 0; a < 2; a++)
#pragma unroll
    for (int c = 0; c < 4; c++) acc[a][c] = (v8f){0, 0, 0, 0, 0, 0, 0, 0};

#if USE_TDM
  const unsigned ldsA = (unsigned)(size_t)(&As[0]);
#endif

  for (int kb = 0; kb < K; kb += 32) {
    if (kb + 32 < K) {  // prefetch next k-tile (global_prefetch_b8)
      __builtin_prefetch(A + (size_t)(mbase + (tid & 63)) * K + kb + 32, 0, 1);
      __builtin_prefetch(Bm + (size_t)(kb + 32 + (tid & 31)) * N + nbase, 0, 1);
    }
    __syncthreads();
#if USE_TDM
    // async DMA of A tile 64x32 (padded to stride 40) while we transpose B
    if (wave == 0)
      tdm_load_2d(ldsA, A + (size_t)mbase * K + kb, G1W0_A, K - kb, M - mbase,
                  32, 64, K);
#else
#pragma unroll
    for (int idx = tid; idx < 256; idx += 128) {
      int r = idx >> 2, c = (idx & 3) * 8;
      *(uint4*)(&As[r * 40 + c]) =
          *(const uint4*)(A + (size_t)(mbase + r) * K + kb + c);
    }
#endif
    // B tile 32x128: 512 uint4, coalesced load + transposed LDS scatter
#pragma unroll
    for (int idx = tid; idx < 512; idx += 128) {
      int r = idx >> 4, c = (idx & 15) * 8;
      U4 t;
      t.q = *(const uint4*)(Bm + (size_t)(kb + r) * N + nbase + c);
#pragma unroll
      for (int s2 = 0; s2 < 8; s2++) BsT[(c + s2) * 40 + r] = t.us[s2];
    }
#if USE_TDM
    if (wave == 0) __builtin_amdgcn_s_wait_tensorcnt(0);
#endif
    __syncthreads();

    FragBF af[2], bfr[4];
#pragma unroll
    for (int ms = 0; ms < 2; ms++) {
      const u16t* p = &As[(wm * 32 + ms * 16 + ln) * 40 + hf * 8];
      af[ms].q[0] = *(const uint4*)(p);       // K = hf*8 .. +7
      af[ms].q[1] = *(const uint4*)(p + 16);  // K = 16+hf*8 .. +7
    }
#pragma unroll
    for (int ns = 0; ns < 4; ns++) {
      const u16t* p = &BsT[(wn * 64 + ns * 16 + ln) * 40 + hf * 8];
      bfr[ns].q[0] = *(const uint4*)(p);      // K = hf*8 .. +7
      bfr[ns].q[1] = *(const uint4*)(p + 16);  // K = 16+hf*8 .. +7
    }
#pragma unroll
    for (int ms = 0; ms < 2; ms++)
#pragma unroll
      for (int ns = 0; ns < 4; ns++)
        acc[ms][ns] = __builtin_amdgcn_wmma_f32_16x16x32_bf16(
            false, af[ms].v, false, bfr[ns].v, (short)0, acc[ms][ns], false,
            false);
  }

#pragma unroll
  for (int ms = 0; ms < 2; ms++) {
#pragma unroll
    for (int ns = 0; ns < 4; ns++) {
      int col = nbase + wn * 64 + ns * 16 + ln;
      float bc = bias ? bias[col] : 0.0f;
#pragma unroll
      for (int i = 0; i < 8; i++) {
        int row = mbase + wm * 32 + ms * 16 + hf * 8 + i;
        float v = acc[ms][ns][i] + bc;
        if (Cbf) Cbf[(size_t)row * N + col] = f2bf(v);
        if (Cf)  Cf[(size_t)row * N + col] = v;
      }
    }
  }
}

// ------------------------------------------------------- flash attention core
// One block = (b, h, 64 q-rows); 4 waves, each owns 16 q-rows.
// K tile: manual transposed stage [w][dk] (stride 72) -> S B-frags 2x b128.
// V tile: TDM DMA [dk][w] (stride 72 via TDM padding) -> O B-frags 2x b128.
// Mask tile: TDM DMA (int32, unpadded).
__global__ __launch_bounds__(128)
void flash_attn(const u16t* __restrict__ Qp, const u16t* __restrict__ Kp,
                const u16t* __restrict__ Vp, const int* __restrict__ mask,
                u16t* __restrict__ X) {
  __shared__ u16t KtT[64 * 72];
  __shared__ u16t Vt[64 * 72];
  __shared__ u16t Pt[4 * 16 * 72];
  __shared__ int  Mt[64 * 64];

  const int tid  = threadIdx.x;
  const int lane = tid & 31;
  const int hf   = lane >> 4;
  const int ln   = lane & 15;
  const int wave = tid >> 5;

  const int qt = blockIdx.x & 15;
  const int h  = (blockIdx.x >> 4) & 15;
  const int b  = blockIdx.x >> 8;

  const int qb = qt * 64 + wave * 16;

  // Q A-fragments: 16 rows x 64 K (2 frags of K=32), straight from global
  FragBF aq[2];
#pragma unroll
  for (int f = 0; f < 2; f++) {
    const u16t* p =
        Qp + (size_t)(b * LQn + qb + ln) * DM + h * 64 + f * 32 + hf * 8;
    aq[f].q[0] = *(const uint4*)(p);
    aq[f].q[1] = *(const uint4*)(p + 16);
  }

  float m8[8], l8[8];
  v8f o[4];
#pragma unroll
  for (int i = 0; i < 8; i++) { m8[i] = -3.0e38f; l8[i] = 0.0f; }
#pragma unroll
  for (int d = 0; d < 4; d++) o[d] = (v8f){0, 0, 0, 0, 0, 0, 0, 0};

#if USE_TDM
  const unsigned ldsV = (unsigned)(size_t)(&Vt[0]);
  const unsigned ldsM = (unsigned)(size_t)(&Mt[0]);
#endif

  for (int wb = 0; wb < HW; wb += 64) {
    if (wb + 64 < HW) {
      __builtin_prefetch(
          Kp + (size_t)(b * DM + h * 64 + (tid & 63)) * HW + wb + 64, 0, 1);
      __builtin_prefetch(
          Vp + (size_t)(b * DM + h * 64 + (tid & 63)) * HW + wb + 64, 0, 1);
    }
    __syncthreads();
#if USE_TDM
    if (wave == 0) {
      tdm_load_2d(ldsV, Vp + (size_t)(b * DM + h * 64) * HW + wb, G1W0_V,
                  HW - wb, 64, 64, 64, HW);
      tdm_load_2d(ldsM, mask + (size_t)(b * LQn + qt * 64) * HW + wb, G1W0_M,
                  HW - wb, 64, 64, 64, HW);
    }
#endif
    // K tile: coalesced load, transposed scatter to LDS [w][dk]
#pragma unroll
    for (int idx = tid; idx < 512; idx += 128) {
      int r = idx >> 3, c = (idx & 7) * 8;  // r = dk, c = w base
      size_t g = (size_t)(b * DM + h * 64 + r) * HW + wb + c;
      U4 t;
      t.q = *(const uint4*)(Kp + g);
#pragma unroll
      for (int s2 = 0; s2 < 8; s2++) KtT[(c + s2) * 72 + r] = t.us[s2];
#if !USE_TDM
      *(uint4*)(&Vt[r * 72 + c]) = *(const uint4*)(Vp + g);
#endif
    }
#if USE_TDM
    if (wave == 0) __builtin_amdgcn_s_wait_tensorcnt(0);
#else
#pragma unroll
    for (int idx = tid; idx < 1024; idx += 128) {
      int r = idx >> 4, c = (idx & 15) * 4;
      *(uint4*)(&Mt[r * 64 + c]) =
          *(const uint4*)(mask + (size_t)(b * LQn + qt * 64 + r) * HW + wb + c);
    }
#endif
    __syncthreads();

    // S = Q @ K_tile / sqrt(dk)   (4 n-subtiles x 2 k-steps)
    v8f s[4];
#pragma unroll
    for (int nt = 0; nt < 4; nt++) {
      v8f acc = (v8f){0, 0, 0, 0, 0, 0, 0, 0};
#pragma unroll
      for (int f = 0; f < 2; f++) {
        FragBF bk;
        const u16t* p = &KtT[(nt * 16 + ln) * 72 + f * 32 + hf * 8];
        bk.q[0] = *(const uint4*)(p);
        bk.q[1] = *(const uint4*)(p + 16);
        acc = __builtin_amdgcn_wmma_f32_16x16x32_bf16(false, aq[f].v, false,
                                                      bk.v, (short)0, acc,
                                                      false, false);
      }
      s[nt] = acc * 0.125f;
    }
    // mask
#pragma unroll
    for (int nt = 0; nt < 4; nt++)
#pragma unroll
      for (int i = 0; i < 8; i++) {
        int ql = wave * 16 + hf * 8 + i;
        if (Mt[ql * 64 + nt * 16 + ln] == 0) s[nt][i] = -1.0e9f;
      }
    // online softmax: row stats uniform across each 16-lane half
#pragma unroll
    for (int i = 0; i < 8; i++) {
      float mx = fmaxf(fmaxf(s[0][i], s[1][i]), fmaxf(s[2][i], s[3][i]));
#pragma unroll
      for (int off = 1; off < 16; off <<= 1)
        mx = fmaxf(mx, __shfl_xor(mx, off, 32));
      float mn   = fmaxf(m8[i], mx);
      float corr = __expf(m8[i] - mn);
      float rs   = 0.0f;
#pragma unroll
      for (int nt = 0; nt < 4; nt++) {
        float p  = __expf(s[nt][i] - mn);
        s[nt][i] = p;
        rs += p;
      }
#pragma unroll
      for (int off = 1; off < 16; off <<= 1) rs += __shfl_xor(rs, off, 32);
      l8[i] = l8[i] * corr + rs;
      m8[i] = mn;
#pragma unroll
      for (int d = 0; d < 4; d++) o[d][i] *= corr;
    }
    // C-layout P -> LDS -> A-layout P (wave-private region, stride 72)
    u16t* pw = &Pt[wave * 16 * 72];
#pragma unroll
    for (int nt = 0; nt < 4; nt++)
#pragma unroll
      for (int i = 0; i < 8; i++)
        pw[(hf * 8 + i) * 72 + nt * 16 + ln] = f2bf(s[nt][i]);
    __syncthreads();  // order LDS store->load (also a convergence point)

    FragBF ap[2];
#pragma unroll
    for (int f = 0; f < 2; f++) {
      const u16t* p = &pw[ln * 72 + f * 32 + hf * 8];
      ap[f].q[0] = *(const uint4*)(p);
      ap[f].q[1] = *(const uint4*)(p + 16);
    }
    // O += P @ V^T  (B[k=w][n=d] = V[d][w] -> contiguous K per lane)
#pragma unroll
    for (int dt = 0; dt < 4; dt++) {
#pragma unroll
      for (int f = 0; f < 2; f++) {
        FragBF bv;
        const u16t* p = &Vt[(dt * 16 + ln) * 72 + f * 32 + hf * 8];
        bv.q[0] = *(const uint4*)(p);
        bv.q[1] = *(const uint4*)(p + 16);
        o[dt] = __builtin_amdgcn_wmma_f32_16x16x32_bf16(
            false, ap[f].v, false, bv.v, (short)0, o[dt], false, false);
      }
    }
  }
  // normalize + scatter to (B, LQ, HEAD*DK) bf16
#pragma unroll
  for (int dt = 0; dt < 4; dt++)
#pragma unroll
    for (int i = 0; i < 8; i++) {
      int row = b * LQn + qb + hf * 8 + i;
      float v = o[dt][i] / l8[i];
      X[(size_t)row * DM + h * 64 + dt * 16 + ln] = f2bf(v);
    }
}

// ---------------------------------------------------------------- launcher
extern "C" void kernel_launch(void* const* d_in, const int* in_sizes, int n_in,
                              void* d_out, int out_size, void* d_ws,
                              size_t ws_size, hipStream_t stream) {
  const float* query = (const float*)d_in[0];
  const float* key   = (const float*)d_in[1];
  const float* value = (const float*)d_in[2];
  const int*   mask  = (const int*)d_in[3];
  const float* Wq    = (const float*)d_in[4];
  const float* bq    = (const float*)d_in[5];
  const float* Wk    = (const float*)d_in[6];
  const float* bk    = (const float*)d_in[7];
  const float* Wv    = (const float*)d_in[8];
  const float* bv    = (const float*)d_in[9];
  float* out = (float*)d_out;

  const int NQ = BB * LQn * DM;  // 4194304
  const int NW = DM * DM;        // 1048576

  u16t* qbf  = (u16t*)d_ws;
  u16t* kbf  = qbf + NQ;
  u16t* vbf  = kbf + NQ;
  u16t* wqbf = vbf + NQ;  // 4 * NW
  u16t* wkbf = wqbf + 4 * NW;
  u16t* wvbf = wkbf + NW;
  u16t* t0   = wvbf + NW;
  u16t* t1   = t0 + NQ;
  u16t* kpbf = t1 + NQ;
  u16t* vpbf = kpbf + NQ;
  u16t* xbf  = vpbf + NQ;

  // f32 -> bf16 staging
  cvt_f32_bf16<<<NQ / 1024, 256, 0, stream>>>(query, qbf, NQ);
  cvt_f32_bf16<<<NQ / 1024, 256, 0, stream>>>(key, kbf, NQ);
  cvt_f32_bf16<<<NQ / 1024, 256, 0, stream>>>(value, vbf, NQ);
  cvt_f32_bf16<<<(4 * NW) / 1024, 256, 0, stream>>>(Wq, wqbf, 4 * NW);
  cvt_f32_bf16<<<NW / 1024, 256, 0, stream>>>(Wk, wkbf, NW);
  cvt_f32_bf16<<<NW / 1024, 256, 0, stream>>>(Wv, wvbf, NW);

  dim3 gg(DM / 128, (BB * LQn) / 64);  // (8, 64)

  // q = (((q W0+b0) W1+b1) W2+b2) W3+b3
  gemm_bf16<<<gg, 128, 0, stream>>>(qbf, wqbf + 0 * NW, bq + 0 * DM, t0,
                                    nullptr, BB * LQn, DM, DM);
  gemm_bf16<<<gg, 128, 0, stream>>>(t0, wqbf + 1 * NW, bq + 1 * DM, t1,
                                    nullptr, BB * LQn, DM, DM);
  gemm_bf16<<<gg, 128, 0, stream>>>(t1, wqbf + 2 * NW, bq + 2 * DM, t0,
                                    nullptr, BB * LQn, DM, DM);
  gemm_bf16<<<gg, 128, 0, stream>>>(t0, wqbf + 3 * NW, bq + 3 * DM, t1,
                                    nullptr, BB * LQn, DM, DM);
  // k/v projections
  gemm_bf16<<<gg, 128, 0, stream>>>(kbf, wkbf, bk, kpbf, nullptr, BB * DM, HW,
                                    HW);
  gemm_bf16<<<gg, 128, 0, stream>>>(vbf, wvbf, bv, vpbf, nullptr, BB * DM, HW,
                                    HW);
  // fused attention (no score materialization)
  flash_attn<<<BB * HEADS * (LQn / 64), 128, 0, stream>>>(t1, kpbf, vpbf, mask,
                                                          xbf);
  // out projection (fp32 result straight to d_out)
  gemm_bf16<<<gg, 128, 0, stream>>>(xbf, wqbf + 3 * NW, bq + 3 * DM, nullptr,
                                    out, BB * LQn, DM, DM);
}
